// SelfAttention_24223615550211
// MI455X (gfx1250) — matrix operand order repaired and checked by reading, not compiled, not run on hardware
//
#include <hip/hip_runtime.h>

// Self-attention (B=4, C=128, N=4096) for gfx1250 / MI455X.
// Pipeline:
//   1) qkv_frag_kernel : Q,K,V = W*x + b, written as f16 WMMA fragments:
//        K -> A-fragment layout (rows j, K-dim = channel) per 16x32 tile
//        Q -> B-fragment layout (rows channel, cols i)
//        V -> B-fragment layout (rows j, cols c) for the AV matmul
//   2) wo_frag_kernel  : Wo^T -> B-fragment layout (rows c, cols o)
//   3) attn_fused_kernel : flash attention (S^T = K x Q, online softmax over j,
//        O = P x V^T) + residual + gamma + output projection (Wo) fused,
//        all matmuls via v_wmma_f32_16x16x32_f16, f32 accumulation.
//        Rescale of the O accumulators is guarded by a wave-uniform branch
//        (rare once the running max stabilizes).

typedef __attribute__((ext_vector_type(16))) _Float16 v16h;
typedef __attribute__((ext_vector_type(8)))  _Float16 v8h;
typedef __attribute__((ext_vector_type(8)))  float    v8f;
typedef __attribute__((ext_vector_type(4)))  float    v4f;

#define WMMA_F16(A, B, C) \
  __builtin_amdgcn_wmma_f32_16x16x32_f16(false, (A), false, (B), (short)0, (C), false, false)

// ---------------------------------------------------------------------------
// Fragment element mappings (wave32, 16-bit data), derived from CDNA5 ISA:
//  A (16x32, M=m rows, K=k):  lane = m + 16*((k%16)/8) ; half = (k&7) + 8*(k/16)
//  B (32x16, K=k rows, N=n):  as A, m->n (V); Q/Wo (and K, sa^T as A) use lane = n + 16*(k/16), half = k%16
//  C/D (16x16 f32):           lane = n + 16*(m/8)      ; reg  = m%8
// ---------------------------------------------------------------------------

__global__ void __launch_bounds__(256) qkv_frag_kernel(
    const float* __restrict__ x,
    const float* __restrict__ Wq, const float* __restrict__ bq,
    const float* __restrict__ Wk, const float* __restrict__ bk,
    const float* __restrict__ Wv, const float* __restrict__ bv,
    _Float16* __restrict__ qfrag,   // [b][it(256)][okt(4)][lane(32)][16]
    _Float16* __restrict__ kfrag,   // [b][jt(256)][okt(4)][lane(32)][16]
    _Float16* __restrict__ vfrag)   // [b][jc(128)][ct(8)][lane(32)][16]
{
  __shared__ float xs[128 * 32];            // x tile [c=128][p=32]
  const int blk = blockIdx.x;               // 512 blocks = B * (N/32)
  const int b   = blk >> 7;
  const int p0  = (blk & 127) * 32;

  for (int idx = threadIdx.x; idx < 128 * 32; idx += 256) {
    const int c = idx >> 5, p = idx & 31;
    xs[idx] = x[(((b << 7) + c) << 12) + p0 + p];
  }
  __syncthreads();

  const int p   = threadIdx.x & 31;         // position within chunk (lane)
  const int og  = threadIdx.x >> 5;         // output-channel group (8 waves)
  const int pos = p0 + p;

  // ---- Q: B-fragment (rows = channel o, cols = i) ----
  for (int oi = 0; oi < 16; ++oi) {
    const int o = og * 16 + oi;
    const float* wr = Wq + o * 128;
    float acc = bq[o];
#pragma unroll 8
    for (int c = 0; c < 128; ++c) acc = fmaf(wr[c], xs[c * 32 + p], acc);
    const int it = pos >> 4, il = pos & 15;
    const int okt = o >> 5, o32 = o & 31;
    const int lane = il + ((o32 >> 4) << 4);
    const int h = o32 & 15;
    qfrag[(((((b << 8) + it) << 2) + okt) * 32 + lane) * 16 + h] = (_Float16)acc;
  }
  // ---- K: A-fragment (rows = j, K-dim = channel o) ----
  for (int oi = 0; oi < 16; ++oi) {
    const int o = og * 16 + oi;
    const float* wr = Wk + o * 128;
    float acc = bk[o];
#pragma unroll 8
    for (int c = 0; c < 128; ++c) acc = fmaf(wr[c], xs[c * 32 + p], acc);
    const int jt = pos >> 4, jl = pos & 15;
    const int okt = o >> 5, o32 = o & 31;
    const int lane = jl + ((o32 >> 4) << 4);
    const int h = o32 & 15;
    kfrag[(((((b << 8) + jt) << 2) + okt) * 32 + lane) * 16 + h] = (_Float16)acc;
  }
  // ---- V: B-fragment (rows = j (K-dim), cols = channel c) ----
  for (int oi = 0; oi < 16; ++oi) {
    const int o = og * 16 + oi;
    const float* wr = Wv + o * 128;
    float acc = bv[o];
#pragma unroll 8
    for (int c = 0; c < 128; ++c) acc = fmaf(wr[c], xs[c * 32 + p], acc);
    const int jc = pos >> 5, j32 = pos & 31;
    const int ct = o >> 4;
    const int lane = (o & 15) + (((j32 & 15) >> 3) << 4);
    const int h = (j32 & 7) + ((j32 >> 4) << 3);
    vfrag[(((((b << 7) + jc) << 3) + ct) * 32 + lane) * 16 + h] = (_Float16)acc;
  }
}

// Wo^T as B-fragments: rows c (K-dim, 4 chunks of 32), cols o (8 tiles of 16)
__global__ void __launch_bounds__(256) wo_frag_kernel(
    const float* __restrict__ Wo, _Float16* __restrict__ wofrag)
{
  const int idx = blockIdx.x * 256 + threadIdx.x;   // 64 blocks -> 16384
  const int o = idx >> 7, c = idx & 127;
  const int ot = o >> 4, ck = c >> 5, c32 = c & 31;
  const int lane = (o & 15) + ((c32 >> 4) << 4);
  const int h = c & 15;
  wofrag[((((ot << 2) + ck) * 32) + lane) * 16 + h] = (_Float16)Wo[idx];
}

__global__ void __launch_bounds__(256) attn_fused_kernel(
    const float* __restrict__ x,
    const float* __restrict__ gamma,
    const float* __restrict__ bo,
    const _Float16* __restrict__ qfrag,
    const _Float16* __restrict__ kfrag,
    const _Float16* __restrict__ vfrag,
    const _Float16* __restrict__ wofrag,
    float* __restrict__ out)
{
  constexpr int LDSROW = 144;                       // halves; 288B, 16B aligned
  __shared__ _Float16 sa16[8 * 16 * LDSROW];
  const int lane = threadIdx.x & 31;
  const int wid  = threadIdx.x >> 5;
  const int tile = blockIdx.x * 8 + wid;            // 128 blocks -> 1024 tiles
  const int b    = tile >> 8;
  const int it   = tile & 255;
  const int half = lane >> 4;                       // 0: lanes 0-15, 1: 16-31
  const int col  = lane & 15;
  _Float16* myLds = sa16 + wid * 16 * LDSROW;

  // Q B-fragments for this i-tile: 4 chunks of 32 channels (kept resident)
  const v16h* qb = (const v16h*)qfrag + (((b << 8) + it) << 2) * 32 + lane;
  const v16h q0 = qb[0], q1 = qb[32], q2 = qb[64], q3 = qb[96];

  float m = -3.0e38f;
  float lsum_half = 0.0f;                           // per-half partial row sum
  v8f oacc[8];                                      // O tile, D(M=i, N=c)
#pragma unroll
  for (int ct = 0; ct < 8; ++ct) oacc[ct] = {};

  const v16h* kb0 = (const v16h*)kfrag + ((b << 8) << 2) * 32 + lane;
  const v16h* vb0 = (const v16h*)vfrag + ((b << 7) << 3) * 32 + lane;

  for (int jc = 0; jc < 128; ++jc) {                // 32 key positions / iter
    const v16h* k0 = kb0 + (2 * jc) * 4 * 32;
    const v16h* k1 = k0 + 4 * 32;
    // S^T tiles: D(M=j, N=i) = K(A) x Q(B), contract over 128 channels
    v8f s0 = {}, s1 = {};
    s0 = WMMA_F16(k0[0],  q0, s0);
    s0 = WMMA_F16(k0[32], q1, s0);
    s0 = WMMA_F16(k0[64], q2, s0);
    s0 = WMMA_F16(k0[96], q3, s0);
    s1 = WMMA_F16(k1[0],  q0, s1);
    s1 = WMMA_F16(k1[32], q1, s1);
    s1 = WMMA_F16(k1[64], q2, s1);
    s1 = WMMA_F16(k1[96], q3, s1);

    // online softmax over j (rows): local regs + one cross-half shuffle
    float vmax = -3.0e38f;
#pragma unroll
    for (int r = 0; r < 8; ++r) {
      vmax = fmaxf(vmax, fmaxf(s0[r], s1[r]));
    }
    vmax = fmaxf(vmax, __shfl_xor(vmax, 16, 32));
    const float mnew = fmaxf(m, vmax);

    float p0[8], p1[8];
#pragma unroll
    for (int r = 0; r < 8; ++r) {
      p0[r] = __expf(s0[r] - mnew);
      p1[r] = __expf(s1[r] - mnew);
    }
    // balanced reduction tree for the local partial sum
    float a0 = (p0[0] + p0[1]) + (p0[2] + p0[3]);
    float a1 = (p0[4] + p0[5]) + (p0[6] + p0[7]);
    float a2 = (p1[0] + p1[1]) + (p1[2] + p1[3]);
    float a3 = (p1[4] + p1[5]) + (p1[6] + p1[7]);
    const float rsum = (a0 + a1) + (a2 + a3);

    // rescale only when the running max actually changed (wave-uniform branch;
    // EXEC stays all-ones for the WMMAs below). corr is identical in both lane
    // halves, so lsum_half can be corrected locally.
    if (__any(vmax > m)) {
      const float corr = __expf(m - mnew);
      float cr[8];
#pragma unroll
      for (int r = 0; r < 8; ++r) cr[r] = __shfl(corr, r + (half << 3), 32);
#pragma unroll
      for (int ct = 0; ct < 8; ++ct) {
#pragma unroll
        for (int r = 0; r < 8; ++r) oacc[ct][r] *= cr[r];
      }
      lsum_half *= corr;
    }
    lsum_half += rsum;
    m = mnew;

    // D(M=j,N=i) pair -> A(M=i, K=j in [0,32)): lane-local repack
    v16h pa = {};
#pragma unroll
    for (int h = 0; h < 8; ++h) {
      pa[h]     = (_Float16)p0[h];
      pa[h + 8] = (_Float16)p1[h];
    }

    // O(i,c) += P(i,j32) x V^T(j32,c): 8 column tiles of 16 channels
    const v16h* vb = vb0 + jc * 8 * 32;
#pragma unroll
    for (int ct = 0; ct < 8; ++ct) oacc[ct] = WMMA_F16(pa, vb[ct * 32], oacc[ct]);
  }

  // combine the two half-lane partial sums once, after the loop
  const float lsum = lsum_half + __shfl_xor(lsum_half, 16, 32);

  // ---- epilogue: sa = gamma * O / lsum + x, staged to LDS as [i][c] f16 ----
  const float g = gamma[0];
  const float invl = 1.0f / lsum;
  float ir[8];
#pragma unroll
  for (int r = 0; r < 8; ++r) ir[r] = __shfl(invl, r + (half << 3), 32) * g;

  const int i0 = it * 16;
#pragma unroll
  for (int ct = 0; ct < 8; ++ct) {
    const int c = ct * 16 + col;                    // this lane's channel
    const float* xp = x + ((long)((b << 7) + c) << 12) + i0 + (half << 3);
    const v4f xlo = *(const v4f*)xp;
    const v4f xhi = *(const v4f*)(xp + 4);
#pragma unroll
    for (int r = 0; r < 8; ++r) {
      const float xr = (r < 4) ? xlo[r] : xhi[r - 4];
      const float sa = oacc[ct][r] * ir[r] + xr;
      const int i = r + (half << 3);
      myLds[i * LDSROW + c] = (_Float16)sa;
    }
  }
  __syncthreads();

  // sa^T A-fragments (M=i, K=c) straight from LDS: 2x b128 per chunk
  v16h af[4];
#pragma unroll
  for (int ck = 0; ck < 4; ++ck) {
    const _Float16* lp = myLds + col * LDSROW + ck * 32 + (half << 4);
    const v8h lo = *(const v8h*)lp;
    const v8h hi = *(const v8h*)(lp + 8);
    af[ck] = __builtin_shufflevector(lo, hi, 0, 1, 2, 3, 4, 5, 6, 7,
                                     8, 9, 10, 11, 12, 13, 14, 15);
  }

  // out(i,o) = sa^T(i,c) x Wo^T(c,o) + bo : 8 output tiles of 16 channels
  const v16h* wb0 = (const v16h*)wofrag + lane;
#pragma unroll
  for (int ot = 0; ot < 8; ++ot) {
    const v16h* wb = wb0 + (ot << 2) * 32;
    v8f pacc = {};
    pacc = WMMA_F16(af[0], wb[0],  pacc);
    pacc = WMMA_F16(af[1], wb[32], pacc);
    pacc = WMMA_F16(af[2], wb[64], pacc);
    pacc = WMMA_F16(af[3], wb[96], pacc);
    const int o = ot * 16 + col;
    const float bov = bo[o];
    float* op = out + ((long)((b << 7) + o) << 12) + i0 + (half << 3);
    v4f r0, r1;
#pragma unroll
    for (int r = 0; r < 4; ++r) { r0[r] = pacc[r] + bov; r1[r] = pacc[r + 4] + bov; }
    *(v4f*)op = r0;
    *(v4f*)(op + 4) = r1;
  }
}

extern "C" void kernel_launch(void* const* d_in, const int* in_sizes, int n_in,
                              void* d_out, int out_size, void* d_ws, size_t ws_size,
                              hipStream_t stream) {
  const float* x     = (const float*)d_in[0];
  const float* Wq    = (const float*)d_in[1];
  const float* bq    = (const float*)d_in[2];
  const float* Wk    = (const float*)d_in[3];
  const float* bk    = (const float*)d_in[4];
  const float* Wv    = (const float*)d_in[5];
  const float* bv    = (const float*)d_in[6];
  const float* gamma = (const float*)d_in[7];
  const float* Wo    = (const float*)d_in[8];
  const float* bo    = (const float*)d_in[9];

  char* ws = (char*)d_ws;
  _Float16* qfrag  = (_Float16*)(ws);                    // 4 MB
  _Float16* kfrag  = (_Float16*)(ws + (4u << 20));       // 4 MB
  _Float16* vfrag  = (_Float16*)(ws + (8u << 20));       // 4 MB
  _Float16* wofrag = (_Float16*)(ws + (12u << 20));      // 32 KB

  qkv_frag_kernel<<<512, 256, 0, stream>>>(x, Wq, bq, Wk, bk, Wv, bv,
                                           qfrag, kfrag, vfrag);
  wo_frag_kernel<<<64, 256, 0, stream>>>(Wo, wofrag);
  attn_fused_kernel<<<128, 256, 0, stream>>>(x, gamma, bo, qfrag, kfrag, vfrag,
                                             wofrag, (float*)d_out);
}
